// GPT_55241869361798
// MI455X (gfx1250) — matrix operand order repaired and checked by reading, not compiled, not run on hardware
//
#include <hip/hip_runtime.h>
#include <math.h>

typedef __attribute__((ext_vector_type(16))) _Float16 v16h;
typedef __attribute__((ext_vector_type(8)))  _Float16 v8h;
typedef __attribute__((ext_vector_type(4)))  _Float16 v4h;
typedef __attribute__((ext_vector_type(8)))  float    v8f;
typedef __attribute__((ext_vector_type(4)))  float    v4f;

#define LNUM 6
#define BBS  4
#define TTS  1024
#define EES  1024
#define HQN  16
#define HKVN 4
#define DDS  64
#define GGS  4
#define EKVS 256
#define FFS  4096
#define VVS  32000
#define MMS  (BBS*TTS)   // 4096 rows

// GEMM block tiling
#define BM 128
#define BN 64
#define BK 32
#define LDA 40   // halves per sA row (32 + 8 pad) -> row base 80B, 16B aligned
#define LDB 40

// ---------------------------------------------------------------------------
// Embedding gather: x[row, :] = wte[idx[row], :]
// ---------------------------------------------------------------------------
__global__ void embed_kernel(const int* __restrict__ idx,
                             const float* __restrict__ wte,
                             float* __restrict__ x) {
    const int row = blockIdx.x;
    const size_t src = (size_t)idx[row] * EES;
    const size_t dst = (size_t)row * EES;
    for (int e = threadIdx.x; e < EES; e += blockDim.x)
        x[dst + e] = wte[src + e];
}

// ---------------------------------------------------------------------------
// LayerNorm over last dim (E=1024). One block (256 threads) per row.
// ---------------------------------------------------------------------------
__global__ void ln_kernel(const float* __restrict__ x,
                          const float* __restrict__ w,
                          const float* __restrict__ b,
                          float* __restrict__ out) {
    __shared__ float rs[256], rs2[256];
    const int tid = threadIdx.x;
    const size_t base = (size_t)blockIdx.x * EES;
    float s = 0.f, s2 = 0.f;
    for (int e = tid; e < EES; e += 256) {
        float v = x[base + e];
        s += v; s2 += v * v;
    }
    rs[tid] = s; rs2[tid] = s2;
    __syncthreads();
    for (int off = 128; off > 0; off >>= 1) {
        if (tid < off) { rs[tid] += rs[tid + off]; rs2[tid] += rs2[tid + off]; }
        __syncthreads();
    }
    const float mean = rs[0] * (1.0f / EES);
    const float var  = rs2[0] * (1.0f / EES) - mean * mean;
    const float rstd = rsqrtf(var + 1e-5f);
    for (int e = tid; e < EES; e += 256)
        out[base + e] = (x[base + e] - mean) * rstd * w[e] + b[e];
}

// ---------------------------------------------------------------------------
// WMMA GEMM:  C[M,N] = act(A[M,K] @ W[N,K]^T + bias) (+ residual)
// Block = 256 threads = 8 waves, block tile 128(M) x 64(N), K-step 32.
// A/W staged via LDS as f16 (converted once per block).
// Waves laid out 4(M) x 2(N); each wave owns a 32x32 register tile =
// 2x2 accumulators -> 4 x v_wmma_f32_16x16x32_f16 per K-step, each A/B
// LDS fragment reused twice.
// Fragment lane mapping per CDNA5 ISA 7.12.2 (wave32):
//   A 16x32: lane&15 = M row; VGPR0-3: K = laneHalf*8+0..7,
//            VGPR4-7: K = 16+laneHalf*8+0..7.
//   B 32x16: lane&15 = N col; same K order as A (VGPR0-3: laneHalf*8+0..7, VGPR4-7: 16+laneHalf*8+0..7).
//   C/D:     VGPR r -> M = r + 8*laneHalf, N = lane&15.
// act: 0 = none, 1 = tanh-GELU.
// ---------------------------------------------------------------------------
__global__ void __launch_bounds__(256)
gemm_kernel(const float* __restrict__ A,
            const float* __restrict__ W,
            const float* __restrict__ bias,
            const float* __restrict__ residual,
            float* __restrict__ C,
            int M, int N, int K, int act) {
    __shared__ _Float16 sA[BM * LDA];   // 10 KB
    __shared__ _Float16 sB[BN * LDB];   //  5 KB

    const int tid      = threadIdx.x;
    const int lane     = tid & 31;
    const int wave     = tid >> 5;
    const int laneHalf = lane >> 4;
    const int lane16   = lane & 15;

    const int blockM = blockIdx.y * BM;
    const int blockN = blockIdx.x * BN;

    // wave layout: 4 x 2
    const int wm = wave >> 1;            // 0..3
    const int wn = wave & 1;             // 0..1
    const int waveM = wm * 32;           // within block tile
    const int waveN = wn * 32;

    // staging assignments
    const int aRow = tid >> 1;                 // 0..127
    const int aKh  = (tid & 1) * 16;           // 0 or 16
    const int bRow = tid & 63;                 // 0..63
    const int bKh  = (tid >> 6) * 8;           // 0,8,16,24

    const float* aSrc = A + (size_t)(blockM + aRow) * K + aKh;
    const float* bSrc = W + (size_t)(blockN + bRow) * K + bKh;

    v8f acc00 = {}, acc01 = {}, acc10 = {}, acc11 = {};

    for (int k0 = 0; k0 < K; k0 += BK) {
        // ---- stage A tile (128x32 f32 -> f16) ----
        {
            __builtin_prefetch(aSrc + k0 + BK, 0, 1);   // global_prefetch_b8
            const v4f* ap = (const v4f*)(aSrc + k0);
            v4f a0 = ap[0], a1 = ap[1], a2 = ap[2], a3 = ap[3];
            v4h h0, h1, h2, h3;
#pragma unroll
            for (int e = 0; e < 4; ++e) {
                h0[e] = (_Float16)a0[e]; h1[e] = (_Float16)a1[e];
                h2[e] = (_Float16)a2[e]; h3[e] = (_Float16)a3[e];
            }
            v4h* dst = (v4h*)&sA[aRow * LDA + aKh];
            dst[0] = h0; dst[1] = h1; dst[2] = h2; dst[3] = h3;
        }
        // ---- stage B tile (64x32 f32 -> f16) ----
        {
            __builtin_prefetch(bSrc + k0 + BK, 0, 1);
            const v4f* bp = (const v4f*)(bSrc + k0);
            v4f b0 = bp[0], b1 = bp[1];
            v4h h0, h1;
#pragma unroll
            for (int e = 0; e < 4; ++e) {
                h0[e] = (_Float16)b0[e]; h1[e] = (_Float16)b1[e];
            }
            v4h* dst = (v4h*)&sB[bRow * LDB + bKh];
            dst[0] = h0; dst[1] = h1;
        }
        __syncthreads();

        // ---- load fragments from LDS ----
        v16h a0, a1, b0, b1;
        {
            const v8h* p0 = (const v8h*)&sA[(waveM + lane16)      * LDA + laneHalf * 8];
            const v8h* p1 = (const v8h*)&sA[(waveM + lane16)      * LDA + 16 + laneHalf * 8];
            const v8h* p2 = (const v8h*)&sA[(waveM + 16 + lane16) * LDA + laneHalf * 8];
            const v8h* p3 = (const v8h*)&sA[(waveM + 16 + lane16) * LDA + 16 + laneHalf * 8];
            v8h lo0 = *p0, hi0 = *p1, lo1 = *p2, hi1 = *p3;
#pragma unroll
            for (int e = 0; e < 8; ++e) {
                a0[e] = lo0[e]; a0[8 + e] = hi0[e];
                a1[e] = lo1[e]; a1[8 + e] = hi1[e];
            }
        }
        {
            const v8h* p0 = (const v8h*)&sB[(waveN + lane16)      * LDB + laneHalf * 8];
            const v8h* p1 = (const v8h*)&sB[(waveN + 16 + lane16) * LDB + laneHalf * 8];
            v8h c0lo = p0[0], c0hi = p0[2], c1lo = p1[0], c1hi = p1[2];
#pragma unroll
            for (int e = 0; e < 8; ++e) {
                b0[e] = c0lo[e]; b0[8 + e] = c0hi[e];
                b1[e] = c1lo[e]; b1[8 + e] = c1hi[e];
            }
        }

        acc00 = __builtin_amdgcn_wmma_f32_16x16x32_f16(false, a0, false, b0, (short)0, acc00, false, false);
        acc01 = __builtin_amdgcn_wmma_f32_16x16x32_f16(false, a0, false, b1, (short)0, acc01, false, false);
        acc10 = __builtin_amdgcn_wmma_f32_16x16x32_f16(false, a1, false, b0, (short)0, acc10, false, false);
        acc11 = __builtin_amdgcn_wmma_f32_16x16x32_f16(false, a1, false, b1, (short)0, acc11, false, false);

        __syncthreads();
    }

    // ---- epilogue: bias / GELU / residual ----
    const float cgelu = 0.7978845608028654f;   // sqrt(2/pi)
    v8f accs[2][2] = {{acc00, acc01}, {acc10, acc11}};
#pragma unroll
    for (int mi = 0; mi < 2; ++mi) {
#pragma unroll
        for (int ni = 0; ni < 2; ++ni) {
            const int col = blockN + waveN + ni * 16 + lane16;
            const float bval = bias ? bias[col] : 0.0f;
#pragma unroll
            for (int r = 0; r < 8; ++r) {
                const int row = blockM + waveM + mi * 16 + r + laneHalf * 8;
                float vout = accs[mi][ni][r] + bval;
                if (act == 1) {
                    const float x3 = vout * vout * vout;
                    vout = 0.5f * vout * (1.0f + tanhf(cgelu * (vout + 0.044715f * x3)));
                }
                const size_t ci = (size_t)row * N + col;
                if (residual) vout += residual[ci];
                C[ci] = vout;
            }
        }
    }
}

// ---------------------------------------------------------------------------
// RoPE in-place. buf layout [B*T, H*D]; grid = (B*T, H), block = D/2 = 32.
// pos is 1-indexed as in the reference (arange(1, T+1)).
// ---------------------------------------------------------------------------
__global__ void rope_kernel(float* __restrict__ buf, int rowStride) {
    const int i = threadIdx.x;                 // 0..31  (D/2)
    const int t = blockIdx.x % TTS;
    const float pos = (float)(t + 1);
    const float theta = __expf(-(2.0f * (float)i / (float)DDS) * 9.210340371976184f); // ln(1e4)
    const float ang = pos * theta;
    const float c = __cosf(ang), s = __sinf(ang);
    float* p = buf + (size_t)blockIdx.x * rowStride + blockIdx.y * DDS;
    const float x1 = p[i];
    const float x2 = p[i + DDS / 2];
    p[i]           = x1 * c - x2 * s;
    p[i + DDS / 2] = x2 * c + x1 * s;
}

// ---------------------------------------------------------------------------
// Causal GQA attention, one block (128 threads) per (b, hq, t) query row.
// q: [B*T, E] (head-major within row), k/v: [B*T, EKV]. y: [B*T, E].
// ---------------------------------------------------------------------------
__global__ void attn_kernel(const float* __restrict__ q,
                            const float* __restrict__ k,
                            const float* __restrict__ v,
                            float* __restrict__ y) {
    __shared__ float sQ[DDS];
    __shared__ float sS[TTS];
    __shared__ float red[128];

    const int tid = threadIdx.x;
    const int t  = blockIdx.x;
    const int hq = blockIdx.y;
    const int b  = blockIdx.z;
    const int kvh = hq / GGS;

    const size_t qoff = ((size_t)(b * TTS + t)) * EES + hq * DDS;
    for (int d = tid; d < DDS; d += 128) sQ[d] = q[qoff + d];
    __syncthreads();

    const float scale = 0.125f;   // 1/sqrt(64)
    for (int j = tid; j <= t; j += 128) {
        const float* krow = k + ((size_t)(b * TTS + j)) * EKVS + kvh * DDS;
        float s = 0.f;
#pragma unroll
        for (int d = 0; d < DDS; ++d) s += sQ[d] * krow[d];
        sS[j] = s * scale;
    }
    __syncthreads();

    // row max
    float m = -1e30f;
    for (int j = tid; j <= t; j += 128) m = fmaxf(m, sS[j]);
    red[tid] = m; __syncthreads();
    for (int off = 64; off > 0; off >>= 1) {
        if (tid < off) red[tid] = fmaxf(red[tid], red[tid + off]);
        __syncthreads();
    }
    m = red[0]; __syncthreads();

    // exp + sum
    float ls = 0.f;
    for (int j = tid; j <= t; j += 128) {
        const float e = __expf(sS[j] - m);
        sS[j] = e; ls += e;
    }
    red[tid] = ls; __syncthreads();
    for (int off = 64; off > 0; off >>= 1) {
        if (tid < off) red[tid] += red[tid + off];
        __syncthreads();
    }
    const float inv = 1.0f / red[0];
    __syncthreads();

    // weighted sum over V: 2 threads per d, split the j range
    const int d    = tid & 63;
    const int half = tid >> 6;
    float acc = 0.f;
    for (int j = half; j <= t; j += 2)
        acc += sS[j] * v[((size_t)(b * TTS + j)) * EKVS + kvh * DDS + d];
    red[tid] = acc; __syncthreads();
    if (tid < DDS)
        y[((size_t)(b * TTS + t)) * EES + hq * DDS + tid] =
            (red[tid] + red[tid + 64]) * inv;
}

// ---------------------------------------------------------------------------
// Host orchestration
// ---------------------------------------------------------------------------
static inline void launch_gemm(const float* A, const float* W, const float* bias,
                               const float* res, float* C,
                               int M, int N, int K, int act, hipStream_t s) {
    dim3 grid(N / BN, M / BM);
    gemm_kernel<<<grid, 256, 0, s>>>(A, W, bias, res, C, M, N, K, act);
}

extern "C" void kernel_launch(void* const* d_in, const int* in_sizes, int n_in,
                              void* d_out, int out_size, void* d_ws, size_t ws_size,
                              hipStream_t stream) {
    const int*   idx    = (const int*)  d_in[0];
    const float* wte    = (const float*)d_in[1];
    const float* Wq     = (const float*)d_in[2];
    const float* bq     = (const float*)d_in[3];
    const float* Wk     = (const float*)d_in[4];
    const float* bk     = (const float*)d_in[5];
    const float* Wv     = (const float*)d_in[6];
    const float* bv     = (const float*)d_in[7];
    const float* Wo     = (const float*)d_in[8];
    const float* bo     = (const float*)d_in[9];
    const float* ln1_w  = (const float*)d_in[10];
    const float* ln1_b  = (const float*)d_in[11];
    const float* ln2_w  = (const float*)d_in[12];
    const float* ln2_b  = (const float*)d_in[13];
    const float* Wfc    = (const float*)d_in[14];
    const float* bfc    = (const float*)d_in[15];
    const float* Wproj  = (const float*)d_in[16];
    const float* bproj  = (const float*)d_in[17];
    const float* lnf_w  = (const float*)d_in[18];
    const float* lnf_b  = (const float*)d_in[19];
    const float* lmhead = (const float*)d_in[20];

    // workspace layout (floats)
    const size_t SZ_X  = (size_t)MMS * EES;    // 4M
    const size_t SZ_KV = (size_t)MMS * EKVS;   // 1M
    float* ws = (float*)d_ws;
    float* x  = ws;
    float* h  = x + SZ_X;
    float* q  = h + SZ_X;
    float* kk = q + SZ_X;
    float* vv = kk + SZ_KV;
    float* y  = vv + SZ_KV;
    float* ff = y + SZ_X;

    embed_kernel<<<MMS, 256, 0, stream>>>(idx, wte, x);

    for (int l = 0; l < LNUM; ++l) {
        const size_t wE2  = (size_t)l * EES * EES;
        const size_t wKV  = (size_t)l * EKVS * EES;
        const size_t wFC  = (size_t)l * FFS * EES;

        ln_kernel<<<MMS, 256, 0, stream>>>(x, ln1_w + l * EES, ln1_b + l * EES, h);

        launch_gemm(h, Wq + wE2, bq + l * EES,  nullptr, q,  MMS, EES,  EES, 0, stream);
        launch_gemm(h, Wk + wKV, bk + l * EKVS, nullptr, kk, MMS, EKVS, EES, 0, stream);
        launch_gemm(h, Wv + wKV, bv + l * EKVS, nullptr, vv, MMS, EKVS, EES, 0, stream);

        rope_kernel<<<dim3(MMS, HQN),  32, 0, stream>>>(q,  EES);
        rope_kernel<<<dim3(MMS, HKVN), 32, 0, stream>>>(kk, EKVS);

        attn_kernel<<<dim3(TTS, HQN, BBS), 128, 0, stream>>>(q, kk, vv, y);

        launch_gemm(y, Wo + wE2, bo + l * EES, x /*residual*/, x, MMS, EES, EES, 0, stream);

        ln_kernel<<<MMS, 256, 0, stream>>>(x, ln2_w + l * EES, ln2_b + l * EES, h);

        launch_gemm(h,  Wfc   + wFC, bfc   + l * FFS, nullptr,        ff, MMS, FFS, EES, 1, stream);
        launch_gemm(ff, Wproj + wFC, bproj + l * EES, x /*residual*/, x,  MMS, EES, FFS, 0, stream);
    }

    ln_kernel<<<MMS, 256, 0, stream>>>(x, lnf_w, lnf_b, h);
    launch_gemm(h, lmhead, nullptr, nullptr, (float*)d_out, MMS, VVS, EES, 0, stream);
}
